// W4A8Linear_68951404970071
// MI455X (gfx1250) — matrix operand order repaired and checked by reading, not compiled, not run on hardware
//
#include <hip/hip_runtime.h>

// ---------------------------------------------------------------------------
// W4A8Linear forward for MI455X (gfx1250, wave32, WMMA).
//
// out[b,o] = sum_i x[b,i] * bf16(weight_q[o,i] * scales[o, i/16])
//   M=32 (batch), K=4096, N=14336. fp32 in, bf16 GEMM, fp32 out.
//
// Memory-bound: ~252 MB of mandatory traffic (fp32 weights dominate) vs
// 3.76 GFLOP => ~11 us at 23.3 TB/s. Strategy: stream weights exactly once,
// dequantize in registers, feed v_wmma_f32_16x16x32_bf16, maximize MLP via
// 4-way K-split (atomic f32 combine) and weight prefetch.
// ---------------------------------------------------------------------------

typedef __attribute__((ext_vector_type(16))) __bf16 bf16x16;
typedef __attribute__((ext_vector_type(8)))  float  f32x8;

#define IN_F     4096
#define OUT_F    14336
#define BATCH    32
#define KSPLIT   4
#define KCHUNK   (IN_F / KSPLIT)          // 1024 K per grid.y slice
#define WAVES_WG 4
#define NT_WG    (WAVES_WG * 16)          // 64 output features per workgroup

// Pack 16 fp32 (scaled by s) into a bf16x16 WMMA operand, in element order.
__device__ __forceinline__ bf16x16 pack_bf16(float4 f0, float4 f1,
                                             float4 f2, float4 f3, float s) {
    bf16x16 r;
    r[0]  = (__bf16)(f0.x * s); r[1]  = (__bf16)(f0.y * s);
    r[2]  = (__bf16)(f0.z * s); r[3]  = (__bf16)(f0.w * s);
    r[4]  = (__bf16)(f1.x * s); r[5]  = (__bf16)(f1.y * s);
    r[6]  = (__bf16)(f1.z * s); r[7]  = (__bf16)(f1.w * s);
    r[8]  = (__bf16)(f2.x * s); r[9]  = (__bf16)(f2.y * s);
    r[10] = (__bf16)(f2.z * s); r[11] = (__bf16)(f2.w * s);
    r[12] = (__bf16)(f3.x * s); r[13] = (__bf16)(f3.y * s);
    r[14] = (__bf16)(f3.z * s); r[15] = (__bf16)(f3.w * s);
    return r;
}

__global__ void w4a8_zero_kernel(float* __restrict__ out, int n) {
    int i = blockIdx.x * blockDim.x + threadIdx.x;
    if (i < n) out[i] = 0.0f;
}

__global__ __launch_bounds__(WAVES_WG * 32)
void w4a8_gemm_kernel(const float* __restrict__ x,        // [32, 4096]
                      const float* __restrict__ wq,       // [14336, 4096]
                      const float* __restrict__ wsc,      // [14336, 256]
                      float* __restrict__ out)            // [32, 14336]
{
    const int tid  = threadIdx.x;
    const int wave = tid >> 5;
    const int lane = tid & 31;
    const int lh   = lane >> 4;    // lane half: K-half selector (B) / K phase (A)
    const int lil  = lane & 15;    // lane-in-half: output col (B) / batch row (A)

    const int o  = blockIdx.x * NT_WG + wave * 16 + lil;  // this lane's out col
    const int k0 = blockIdx.y * KCHUNK;                   // K slice start

    // B operand: lane reads wq[o, k .. k+15] (lanes 0-15: k = k0+32t,
    // lanes 16-31: k = k0+32t+16) -- 64B contiguous, one scale block each.
    const float* wp  = wq  + (size_t)o * IN_F + k0 + lh * 16;
    const float* sp  = wsc + (size_t)o * (IN_F / 16) + (k0 >> 4) + lh;

    // A operand (x, bf16): lane row m=lil; elems 0-15 at k0+lh*16 (contiguous).
    const float* xp0 = x + (size_t)lil * IN_F + k0 + lh * 16;   // batch 0-15
    const float* xp1 = xp0 + (size_t)16 * IN_F;                 // batch 16-31

    f32x8 acc0 = {};  // D tile: batch rows 0-15  x 16 output features
    f32x8 acc1 = {};  // D tile: batch rows 16-31 x 16 output features

#pragma unroll 2
    for (int it = 0; it < KCHUNK / 32; ++it) {
        // Stay ~8 iterations (1 KB/lane) ahead on the weight stream.
        __builtin_prefetch((const void*)(wp + 8 * 32), 0, 0);

        // --- B: load 16 fp32 weights + 1 scale, dequant -> bf16 ------------
        const float4* w4 = (const float4*)wp;
        float4 b0 = w4[0], b1 = w4[1], b2 = w4[2], b3 = w4[3];
        float  s  = *sp;
        bf16x16 bm = pack_bf16(b0, b1, b2, b3, s);

        // --- A: two batch tiles, 16 fp32 each (8 @ +0, 8 @ +8) -------------
        const float4* a4lo = (const float4*)xp0;
        const float4* a4hi = (const float4*)(xp0 + 8);
        bf16x16 am0 = pack_bf16(a4lo[0], a4lo[1], a4hi[0], a4hi[1], 1.0f);

        const float4* c4lo = (const float4*)xp1;
        const float4* c4hi = (const float4*)(xp1 + 8);
        bf16x16 am1 = pack_bf16(c4lo[0], c4lo[1], c4hi[0], c4hi[1], 1.0f);

        // --- 2x v_wmma_f32_16x16x32_bf16 (shared B) ------------------------
        acc0 = __builtin_amdgcn_wmma_f32_16x16x32_bf16(
                   false, am0, false, bm, (short)0, acc0, false, false);
        acc1 = __builtin_amdgcn_wmma_f32_16x16x32_bf16(
                   false, am1, false, bm, (short)0, acc1, false, false);

        wp  += 32;
        sp  += 2;
        xp0 += 32;
        xp1 += 32;
    }

    // D layout: VGPR r -> row r (lanes 0-15) / row r+8 (lanes 16-31), col=lil.
    // Combine the KSPLIT partial sums with f32 atomics.
#pragma unroll
    for (int r = 0; r < 8; ++r) {
        int brow = r + lh * 8;
        atomicAdd(out + (size_t)brow        * OUT_F + o, acc0[r]);
        atomicAdd(out + (size_t)(brow + 16) * OUT_F + o, acc1[r]);
    }
}

extern "C" void kernel_launch(void* const* d_in, const int* in_sizes, int n_in,
                              void* d_out, int out_size, void* d_ws, size_t ws_size,
                              hipStream_t stream) {
    const float* x   = (const float*)d_in[0];   // [32,1,4096]  fp32
    const float* wq  = (const float*)d_in[1];   // [14336,4096] fp32 (fp4 grid)
    const float* wsc = (const float*)d_in[2];   // [14336,256]  fp32 scales
    float* out = (float*)d_out;                 // [32,1,14336] fp32

    // Zero the output (harness poisons it; K-split partials accumulate).
    w4a8_zero_kernel<<<(out_size + 255) / 256, 256, 0, stream>>>(out, out_size);

    dim3 grid(OUT_F / NT_WG, KSPLIT);           // (224, 4)
    dim3 block(WAVES_WG * 32);                  // 128 threads = 4 waves
    w4a8_gemm_kernel<<<grid, block, 0, stream>>>(x, wq, wsc, out);
}
